// GraphClassifier_7164005450106
// MI455X (gfx1250) — matrix operand order repaired and checked by reading, not compiled, not run on hardware
//
#include <hip/hip_runtime.h>
#include <math.h>

// ---------------------------------------------------------------------------
// Types for CDNA5 WMMA (wave32): D(16x16 f32) = A(16x32 bf16) x B(32x16 bf16) + C
// ---------------------------------------------------------------------------
typedef __bf16 v16bf __attribute__((ext_vector_type(16)));
typedef float  v8f   __attribute__((ext_vector_type(8)));

union FragB { v16bf v; uint4 u[2]; };

__device__ __forceinline__ unsigned short f2bf(float f) {
  __bf16 b = (__bf16)f;
  return __builtin_bit_cast(unsigned short, b);
}

__device__ __forceinline__ v8f vzero8() {
  v8f r;
#pragma unroll
  for (int i = 0; i < 8; ++i) r[i] = 0.f;
  return r;
}

__device__ __forceinline__ v8f vbcast8(float f) {
  v8f r;
#pragma unroll
  for (int i = 0; i < 8; ++i) r[i] = f;
  return r;
}

__device__ __forceinline__ v8f wmma_bf(const FragB& a, const FragB& b, v8f c) {
  // v_wmma_f32_16x16x32_bf16
  return __builtin_amdgcn_wmma_f32_16x16x32_bf16(false, a.v, false, b.v,
                                                 (short)0, c, false, false);
}

__device__ __forceinline__ float sigm(float x) { return 1.f / (1.f + __expf(-x)); }

// Async copy of one 16-byte chunk into LDS (per-lane), tracked by ASYNCcnt.
// lds_off: LDS byte offset; g: global source address.
__device__ __forceinline__ void async_cp16(unsigned lds_off, const void* g) {
  asm volatile("global_load_async_to_lds_b128 %0, %1, off"
               :: "v"(lds_off), "v"(g) : "memory");
}
__device__ __forceinline__ void async_wait0() {
  asm volatile("s_wait_asynccnt 0x0" ::: "memory");
}

// ---------------------------------------------------------------------------
// Elementwise prep kernels
// ---------------------------------------------------------------------------
__global__ void k_f32_to_bf16(const float* __restrict__ in,
                              unsigned short* __restrict__ out, long n) {
  long i = (long)blockIdx.x * blockDim.x + threadIdx.x;
  if (i < n) out[i] = f2bf(in[i]);
}

__global__ void k_fill_f32(float* __restrict__ p, float v, long n) {
  long i = (long)blockIdx.x * blockDim.x + threadIdx.x;
  if (i < n) p[i] = v;
}

// Swizzle an f32 weight matrix into WMMA B-fragment order (bf16).
// Frag f = nt*kchunks + kc holds a 32x16 tile; lane l stores its 16 contiguous
// bf16 so GEMM-side loads are 2x b128.
__global__ void k_swizzle_b(const float* __restrict__ W,
                            unsigned short* __restrict__ out,
                            int Kdim, int Ndim, int transposed, int total) {
  int tid = blockIdx.x * blockDim.x + threadIdx.x;
  if (tid >= total) return;
  int f = tid >> 9;          // fragment index
  int r = tid & 511;
  int lane = r >> 4;
  int e = r & 15;
  int kchunks = Kdim >> 5;
  int kc = f % kchunks;
  int nt = f / kchunks;
  int n = nt * 16 + (lane & 15);
  int k = kc * 32 + (e >> 3) * 16 + (lane >> 4) * 8 + (e & 7);
  float v = transposed ? W[(size_t)n * Kdim + k]   // B = W^T, W is (Ndim,Kdim)
                       : W[(size_t)k * Ndim + n];  // B = W,   W is (Kdim,Ndim)
  out[tid] = f2bf(v);
}

// ---------------------------------------------------------------------------
// Fused embed + bidirectional GRU. One wave = one 16-node tile, one direction
// per blockIdx.y. Per-direction weights (Wi||Wh fragments, 192 KB bf16) are
// async-staged into LDS once per block; h state lives in C-layout f32 regs and
// is transposed to A-layout bf16 through a per-wave LDS slab each step.
//
// Gate accumulators are fused: r and z gates need only (x@Wi + h@Wh + bi + bh),
// so both WMMA sequences chain into one accumulator seeded with the bias sum;
// only the n gate keeps its input/hidden halves separate (tanh(in + r*hn)).
// 4 accumulators instead of 6 => lower VGPR pressure, same 24 WMMAs per tile.
//
// Dynamic LDS layout (bytes):
//   [0,      98304)  Wi fragments (24 nt x 4 kc x 512 bf16)
//   [98304, 196608)  Wh fragments
//   [196608,229376)  per-wave 16x128 bf16 h-transpose stage (8 x 4 KB)
// ---------------------------------------------------------------------------
#define GRU_WAVES 8

__global__ __launch_bounds__(256, 1) void k_gru(
    const int* __restrict__ xtok,                 // N x 32 token ids
    const unsigned short* __restrict__ emb,       // V x 128 bf16
    const unsigned short* __restrict__ wi_f, const unsigned short* __restrict__ wh_f,
    const float* __restrict__ bi_f, const float* __restrict__ bh_f,
    const unsigned short* __restrict__ wi_b, const unsigned short* __restrict__ wh_b,
    const float* __restrict__ bi_b, const float* __restrict__ bh_b,
    unsigned short* __restrict__ hcat,            // N x 256 bf16 output
    int N) {
  extern __shared__ unsigned short lds[];

  const int wave = threadIdx.x >> 5;
  const int lane = threadIdx.x & 31;
  const int half = lane >> 4;
  const int mr   = lane & 15;
  const int dir  = blockIdx.y;

  const unsigned short* wi = dir ? wi_b : wi_f;
  const unsigned short* wh = dir ? wh_b : wh_f;
  const float* bi = dir ? bi_b : bi_f;
  const float* bh = dir ? bh_b : bh_f;

  // Stage this direction's weights into LDS (async, 16 B per lane-item).
  // 98304 B each => 6144 items => 24 iterations over 256 threads.
  for (int i = threadIdx.x; i < 6144; i += 256)
    async_cp16((unsigned)i * 16u, (const char*)wi + (size_t)i * 16);
  for (int i = threadIdx.x; i < 6144; i += 256)
    async_cp16(98304u + (unsigned)i * 16u, (const char*)wh + (size_t)i * 16);
  async_wait0();
  __syncthreads();

  const unsigned short* wiL = lds;            // ushort index
  const unsigned short* whL = lds + 49152;
  unsigned short*       stg = lds + 98304 + wave * 2048;

  const int row0 = (blockIdx.x * GRU_WAVES + wave) * 16;
  const int node = min(row0 + mr, N - 1);

  v8f hc[8];
#pragma unroll
  for (int j = 0; j < 8; ++j) hc[j] = vzero8();

  for (int t = 0; t < 32; ++t) {
    const int te = dir ? (31 - t) : t;
    const int tok = xtok[node * 32 + te];
    const unsigned short* xr = emb + (size_t)tok * 128;
    if (t + 1 < 32) {  // prefetch next step's embedding row (global_prefetch_b8)
      const int tn = dir ? (30 - t) : (t + 1);
      __builtin_prefetch(emb + (size_t)xtok[node * 32 + tn] * 128, 0, 0);
    }

    // A fragments of x_t (16x128 bf16): documented 16-bit A interleave.
    FragB axf[4];
#pragma unroll
    for (int kc = 0; kc < 4; ++kc) {
      axf[kc].u[0] = *(const uint4*)(xr + kc * 32 + half * 8);
      axf[kc].u[1] = *(const uint4*)(xr + kc * 32 + 16 + half * 8);
    }

    // C-layout h -> row-major bf16 in LDS -> A-layout fragments.
#pragma unroll
    for (int j = 0; j < 8; ++j)
#pragma unroll
      for (int v = 0; v < 8; ++v)
        stg[(v + 8 * half) * 128 + j * 16 + mr] = f2bf(hc[j][v]);
    __syncthreads();
    FragB ahf[4];
#pragma unroll
    for (int kc = 0; kc < 4; ++kc) {
      const unsigned short* hrp = stg + mr * 128 + kc * 32;
      ahf[kc].u[0] = *(const uint4*)(hrp + half * 8);
      ahf[kc].u[1] = *(const uint4*)(hrp + 16 + half * 8);
    }
    __syncthreads();

    // Gates per 16-wide output tile j (H=128 -> 8 tiles). B frags from LDS.
    for (int j = 0; j < 8; ++j) {
      const int n0 = j * 16 + mr;
      v8f accr = vbcast8(bi[n0] + bh[n0]);              // r preactivation
      v8f accz = vbcast8(bi[128 + n0] + bh[128 + n0]);  // z preactivation
      v8f acni = vbcast8(bi[256 + n0]);                 // n gate, input half
      v8f acnh = vbcast8(bh[256 + n0]);                 // n gate, hidden half
#pragma unroll
      for (int kc = 0; kc < 4; ++kc) {
        FragB b0, b1, b2, b3, b4, b5;
        const unsigned short* p;
        p = wiL + ((j)      * 4 + kc) * 512 + lane * 16; b0.u[0] = *(const uint4*)p; b0.u[1] = *(const uint4*)(p + 8);
        p = whL + ((j)      * 4 + kc) * 512 + lane * 16; b1.u[0] = *(const uint4*)p; b1.u[1] = *(const uint4*)(p + 8);
        p = wiL + ((j + 8)  * 4 + kc) * 512 + lane * 16; b2.u[0] = *(const uint4*)p; b2.u[1] = *(const uint4*)(p + 8);
        p = whL + ((j + 8)  * 4 + kc) * 512 + lane * 16; b3.u[0] = *(const uint4*)p; b3.u[1] = *(const uint4*)(p + 8);
        p = wiL + ((j + 16) * 4 + kc) * 512 + lane * 16; b4.u[0] = *(const uint4*)p; b4.u[1] = *(const uint4*)(p + 8);
        p = whL + ((j + 16) * 4 + kc) * 512 + lane * 16; b5.u[0] = *(const uint4*)p; b5.u[1] = *(const uint4*)(p + 8);
        accr = wmma_bf(axf[kc], b0, accr);
        accr = wmma_bf(ahf[kc], b1, accr);
        accz = wmma_bf(axf[kc], b2, accz);
        accz = wmma_bf(ahf[kc], b3, accz);
        acni = wmma_bf(axf[kc], b4, acni);
        acnh = wmma_bf(ahf[kc], b5, acnh);
      }
#pragma unroll
      for (int v = 0; v < 8; ++v) {
        float r = sigm(accr[v]);
        float z = sigm(accz[v]);
        float nn = tanhf(acni[v] + r * acnh[v]);
        hc[j][v] = (1.f - z) * nn + z * hc[j][v];
      }
    }
  }

  // Store final h (C layout) to hcat[node][dir*128 + col] as bf16.
#pragma unroll
  for (int j = 0; j < 8; ++j)
#pragma unroll
    for (int v = 0; v < 8; ++v) {
      const int nm = min(row0 + v + 8 * half, N - 1);
      hcat[(size_t)nm * 256 + dir * 128 + j * 16 + mr] = f2bf(hc[j][v]);
    }
}

// ---------------------------------------------------------------------------
// Generic bf16 GEMM: (N x Kdim row-major bf16) @ (pre-swizzled B frags) ->
// N x 128 f32. B fragments (shared by all waves) async-staged into LDS.
// ---------------------------------------------------------------------------
__global__ __launch_bounds__(256) void k_gemm_bf16(
    const unsigned short* __restrict__ A, const unsigned short* __restrict__ Bf,
    float* __restrict__ C, int N, int Kdim) {
  extern __shared__ unsigned short lds[];
  const int wave = threadIdx.x >> 5;
  const int lane = threadIdx.x & 31;
  const int half = lane >> 4;
  const int mr   = lane & 15;
  const int row0 = (blockIdx.x * 8 + wave) * 16;
  const int node = min(row0 + mr, N - 1);
  const int kchunks = Kdim >> 5;

  // Stage B fragments: 8 nt * kchunks * 1024 B = kchunks*8192 B.
  const int items = kchunks * 512;  // 16-byte chunks
  for (int i = threadIdx.x; i < items; i += 256)
    async_cp16((unsigned)i * 16u, (const char*)Bf + (size_t)i * 16);
  async_wait0();
  __syncthreads();

  v8f acc[8];
#pragma unroll
  for (int nt = 0; nt < 8; ++nt) acc[nt] = vzero8();

  const unsigned short* ar = A + (size_t)node * Kdim;
  for (int kc = 0; kc < kchunks; ++kc) {
    FragB a;
    a.u[0] = *(const uint4*)(ar + kc * 32 + half * 8);
    a.u[1] = *(const uint4*)(ar + kc * 32 + 16 + half * 8);
#pragma unroll
    for (int nt = 0; nt < 8; ++nt) {
      FragB b;
      const unsigned short* bp = lds + (nt * kchunks + kc) * 512 + lane * 16;
      b.u[0] = *(const uint4*)bp;
      b.u[1] = *(const uint4*)(bp + 8);
      acc[nt] = wmma_bf(a, b, acc[nt]);
    }
  }
#pragma unroll
  for (int nt = 0; nt < 8; ++nt)
#pragma unroll
    for (int v = 0; v < 8; ++v) {
      const int nm = min(row0 + v + 8 * half, N - 1);
      C[(size_t)nm * 128 + nt * 16 + mr] = acc[nt][v];
    }
}

// ---------------------------------------------------------------------------
// GCN scatter machinery
// ---------------------------------------------------------------------------
__global__ void k_deg_count(const int* __restrict__ ei, int NE, float* __restrict__ deg) {
  int e = blockIdx.x * blockDim.x + threadIdx.x;
  if (e < NE) atomicAdd(&deg[ei[NE + e]], 1.0f);
}

__global__ void k_dinv(float* __restrict__ d, int N) {
  int i = blockIdx.x * blockDim.x + threadIdx.x;
  if (i < N) d[i] = rsqrtf(d[i]);  // deg >= 1 (self loop)
}

// out = bias + self-loop message (h_lin[i] * dinv[i]^2)
__global__ void k_gcn_init(const float* __restrict__ hlin, const float* __restrict__ dinv,
                           const float* __restrict__ bias, float* __restrict__ out, int N) {
  long tid = (long)blockIdx.x * blockDim.x + threadIdx.x;
  if (tid >= (long)N * 128) return;
  int i = (int)(tid >> 7), c = (int)(tid & 127);
  float di = dinv[i];
  out[tid] = bias[c] + hlin[tid] * di * di;
}

__global__ void k_gcn_scatter(const int* __restrict__ ei, int NE,
                              const float* __restrict__ dinv,
                              const float* __restrict__ hlin, float* __restrict__ out) {
  long tid = (long)blockIdx.x * blockDim.x + threadIdx.x;
  if (tid >= (long)NE * 32) return;
  int e = (int)(tid >> 5), g = (int)(tid & 31);
  int s = ei[e], d = ei[NE + e];
  float w = dinv[s] * dinv[d];
  const float4 v = *(const float4*)&hlin[(size_t)s * 128 + g * 4];
  float* o = &out[(size_t)d * 128 + g * 4];
  atomicAdd(o + 0, v.x * w);
  atomicAdd(o + 1, v.y * w);
  atomicAdd(o + 2, v.z * w);
  atomicAdd(o + 3, v.w * w);
}

__global__ void k_relu_bf16(const float* __restrict__ in, unsigned short* __restrict__ out, long n) {
  long i = (long)blockIdx.x * blockDim.x + threadIdx.x;
  if (i < n) out[i] = f2bf(fmaxf(in[i], 0.f));
}

// ---------------------------------------------------------------------------
// Pooling + FC
// ---------------------------------------------------------------------------
__global__ void k_pool_cnt(const int* __restrict__ batch, float* __restrict__ cnt, int N) {
  int i = blockIdx.x * blockDim.x + threadIdx.x;
  if (i < N) atomicAdd(&cnt[batch[i]], 1.0f);
}

__global__ void k_pool_sum(const float* __restrict__ out2, const int* __restrict__ batch,
                           float* __restrict__ pooled, int N) {
  long tid = (long)blockIdx.x * blockDim.x + threadIdx.x;
  if (tid >= (long)N * 32) return;
  int i = (int)(tid >> 5), g = (int)(tid & 31);
  int b = batch[i];
  const float4 v = *(const float4*)&out2[(size_t)i * 128 + g * 4];
  float* p = &pooled[(size_t)b * 128 + g * 4];
  atomicAdd(p + 0, fmaxf(v.x, 0.f));
  atomicAdd(p + 1, fmaxf(v.y, 0.f));
  atomicAdd(p + 2, fmaxf(v.z, 0.f));
  atomicAdd(p + 3, fmaxf(v.w, 0.f));
}

__global__ void k_fc(const float* __restrict__ pooled, const float* __restrict__ cnt,
                     const float* __restrict__ Wfc, const float* __restrict__ bfc,
                     float* __restrict__ out) {
  int tid = threadIdx.x;
  if (tid >= 64 * 8) return;
  int g = tid >> 3, c = tid & 7;
  float s = 0.f;
  for (int k = 0; k < 128; ++k) s += pooled[g * 128 + k] * Wfc[k * 8 + c];
  float ct = cnt[g];
  if (ct < 1.f) ct = 1.f;
  out[g * 8 + c] = s / ct + bfc[c];
}

// ---------------------------------------------------------------------------
// Host launcher
// ---------------------------------------------------------------------------
static inline int cdiv(long a, long b) { return (int)((a + b - 1) / b); }

extern "C" void kernel_launch(void* const* d_in, const int* in_sizes, int n_in,
                              void* d_out, int out_size, void* d_ws, size_t ws_size,
                              hipStream_t stream) {
  const int*   x     = (const int*)d_in[0];
  const int*   ei    = (const int*)d_in[1];
  const int*   batch = (const int*)d_in[2];
  const float* emb   = (const float*)d_in[3];
  const float* Wi_f  = (const float*)d_in[4];
  const float* Wh_f  = (const float*)d_in[5];
  const float* bi_f  = (const float*)d_in[6];
  const float* bh_f  = (const float*)d_in[7];
  const float* Wi_b  = (const float*)d_in[8];
  const float* Wh_b  = (const float*)d_in[9];
  const float* bi_b  = (const float*)d_in[10];
  const float* bh_b  = (const float*)d_in[11];
  const float* W1    = (const float*)d_in[12];
  const float* b1    = (const float*)d_in[13];
  const float* W2    = (const float*)d_in[14];
  const float* b2    = (const float*)d_in[15];
  const float* Wfc   = (const float*)d_in[16];
  const float* bfc   = (const float*)d_in[17];

  const int N  = in_sizes[0] / 32;   // 20000
  const int NE = in_sizes[1] / 2;    // 640000
  const long VE = in_sizes[3];       // 32000*128

  // Workspace carve-out (256-byte aligned cursor allocator).
  char* ws = (char*)d_ws;
  size_t cur = 0;
  auto alloc = [&](size_t bytes) -> char* {
    char* p = ws + cur;
    cur += (bytes + 255) & ~(size_t)255;
    return p;
  };
  unsigned short* emb_bf  = (unsigned short*)alloc(VE * 2);
  unsigned short* wif_fr  = (unsigned short*)alloc(384 * 128 * 2);
  unsigned short* whf_fr  = (unsigned short*)alloc(384 * 128 * 2);
  unsigned short* wib_fr  = (unsigned short*)alloc(384 * 128 * 2);
  unsigned short* whb_fr  = (unsigned short*)alloc(384 * 128 * 2);
  unsigned short* w1_fr   = (unsigned short*)alloc(256 * 128 * 2);
  unsigned short* w2_fr   = (unsigned short*)alloc(128 * 128 * 2);
  unsigned short* hcat    = (unsigned short*)alloc((size_t)N * 256 * 2);
  float*          bufA    = (float*)alloc((size_t)N * 128 * 4);  // h_lin (both layers)
  float*          bufB    = (float*)alloc((size_t)N * 128 * 4);  // scatter accumulator
  unsigned short* g1_bf   = (unsigned short*)alloc((size_t)N * 128 * 2);
  float*          dinv    = (float*)alloc((size_t)N * 4);
  float*          pooled  = (float*)alloc(64 * 128 * 4);
  float*          cnt     = (float*)alloc(64 * 4);

  const int TB = 256;

  // 1. Prep: embedding + weights -> bf16 (weights in WMMA B-fragment order).
  k_f32_to_bf16<<<cdiv(VE, TB), TB, 0, stream>>>(emb, emb_bf, VE);
  k_swizzle_b<<<cdiv(384 * 128, TB), TB, 0, stream>>>(Wi_f, wif_fr, 128, 384, 1, 384 * 128);
  k_swizzle_b<<<cdiv(384 * 128, TB), TB, 0, stream>>>(Wh_f, whf_fr, 128, 384, 1, 384 * 128);
  k_swizzle_b<<<cdiv(384 * 128, TB), TB, 0, stream>>>(Wi_b, wib_fr, 128, 384, 1, 384 * 128);
  k_swizzle_b<<<cdiv(384 * 128, TB), TB, 0, stream>>>(Wh_b, whb_fr, 128, 384, 1, 384 * 128);
  k_swizzle_b<<<cdiv(256 * 128, TB), TB, 0, stream>>>(W1, w1_fr, 256, 128, 0, 256 * 128);
  k_swizzle_b<<<cdiv(128 * 128, TB), TB, 0, stream>>>(W2, w2_fr, 128, 128, 0, 128 * 128);

  // 2. Fused embed + bi-GRU (blockIdx.y = direction). 224 KB dynamic LDS:
  //    192 KB staged weights + 32 KB h-transpose stage (<= 320 KB/WGP).
  dim3 ggrid(cdiv(N, 128), 2);
  k_gru<<<ggrid, TB, 229376, stream>>>(x, emb_bf, wif_fr, whf_fr, bi_f, bh_f,
                                       wib_fr, whb_fr, bi_b, bh_b, hcat, N);

  // 3. Degree normalization: deg = 1 (self loop) + in-degree; dinv = deg^-1/2.
  k_fill_f32<<<cdiv(N, TB), TB, 0, stream>>>(dinv, 1.0f, N);
  k_deg_count<<<cdiv(NE, TB), TB, 0, stream>>>(ei, NE, dinv);
  k_dinv<<<cdiv(N, TB), TB, 0, stream>>>(dinv, N);

  // 4. GCN layer 1: h_lin = hcat @ W1 (WMMA), init bias+self-loop, edge scatter.
  k_gemm_bf16<<<cdiv(N, 128), TB, 65536, stream>>>(hcat, w1_fr, bufA, N, 256);
  k_gcn_init<<<cdiv((long)N * 128, TB), TB, 0, stream>>>(bufA, dinv, b1, bufB, N);
  k_gcn_scatter<<<cdiv((long)NE * 32, TB), TB, 0, stream>>>(ei, NE, dinv, bufA, bufB);
  k_relu_bf16<<<cdiv((long)N * 128, TB), TB, 0, stream>>>(bufB, g1_bf, (long)N * 128);

  // 5. GCN layer 2.
  k_gemm_bf16<<<cdiv(N, 128), TB, 32768, stream>>>(g1_bf, w2_fr, bufA, N, 128);
  k_gcn_init<<<cdiv((long)N * 128, TB), TB, 0, stream>>>(bufA, dinv, b2, bufB, N);
  k_gcn_scatter<<<cdiv((long)NE * 32, TB), TB, 0, stream>>>(ei, NE, dinv, bufA, bufB);

  // 6. Mean-pool per graph (relu fused into sum) + FC head.
  k_fill_f32<<<cdiv(64 * 128, TB), TB, 0, stream>>>(pooled, 0.0f, 64 * 128);
  k_fill_f32<<<1, 64, 0, stream>>>(cnt, 0.0f, 64);
  k_pool_cnt<<<cdiv(N, TB), TB, 0, stream>>>(batch, cnt, N);
  k_pool_sum<<<cdiv((long)N * 32, TB), TB, 0, stream>>>(bufB, batch, pooled, N);
  k_fc<<<1, 512, 0, stream>>>(pooled, cnt, Wfc, bfc, (float*)d_out);
}
